// dslDecoder_54992761258144
// MI455X (gfx1250) — matrix operand order repaired and checked by reading, not compiled, not run on hardware
//
#include <hip/hip_runtime.h>
#include <hip/hip_bf16.h>
#include <math.h>

typedef __attribute__((ext_vector_type(16))) _Float16 v16h;
typedef __attribute__((ext_vector_type(8)))  _Float16 v8h;
typedef __attribute__((ext_vector_type(8)))  float    v8f;

#define Bb 64
#define Tt 128
#define Hh 512
#define Ff 16
#define NR 8192
#define TLc 464

// ---------------------------------------------------------------------------
// WMMA GEMM, all-f16 operands: C[z] = act(A[z] @ B[z]^T + bias[z])
//  - A: f16 [M][lda] row-major (K contiguous); rows optionally gathered
//  - BT: f16 [N][ldb] n-major (K contiguous, pre-transposed + zero-padded)
//  - K must be a multiple of 32 (buffers are zero-padded to guarantee this)
//  - epilogue writes f32 C and/or f16 C16 mirror
//  - staging is a pure b128 copy (clamped addresses; OOB results masked at
//    writeout), register-prefetched + LDS double-buffered, 1 barrier/K-step
// ---------------------------------------------------------------------------
__global__ __launch_bounds__(128) void k_gemm16(
    const _Float16* __restrict__ A, long lda, long aF, long aP,
    const _Float16* __restrict__ BT, long ldb, long bF, long bP,
    const float* __restrict__ bias, long biF, long biP,
    float* __restrict__ C, long ldc, long cF, long cP,
    _Float16* __restrict__ C16, long ldc16, long c16F, long c16P,
    int M, int N, int K, int act, int nF,
    const int* __restrict__ bcnt, const int* __restrict__ bbase,
    const int* __restrict__ gidx)
{
  int z = blockIdx.z;
  int f = z % nF, p = z / nF;
  long aOff   = (long)f * aF   + (long)p * aP;
  long bOff   = (long)f * bF   + (long)p * bP;
  long biOff  = (long)f * biF  + (long)p * biP;
  long cOff   = (long)f * cF   + (long)p * cP;
  long c16Off = (long)f * c16F + (long)p * c16P;

  int Meff = M, rowOff = 0;
  if (bcnt) { Meff = bcnt[f]; rowOff = bbase[f]; }

  int m0 = blockIdx.y * 64;
  if (m0 >= Meff) return;
  int n0 = blockIdx.x * 64;

  __shared__ __align__(16) _Float16 As[2][64][40];   // 64 rows x 32 K (+pad)
  __shared__ __align__(16) _Float16 Bs[2][64][40];   // 64 cols(n) x 32 K (+pad)

  int tid  = threadIdx.x;
  int lane = tid & 31;
  int w    = tid >> 5;
  int wm   = (w >> 1) * 32;
  int wn   = (w & 1) * 32;

  // ---- per-thread staging pointers: 2 chunks of 8 halves each, clamped ----
  const _Float16* aPtr[2]; int aR[2], aC8[2];
  #pragma unroll
  for (int i = 0; i < 2; ++i) {
    int ch = tid + i * 128;        // 0..255
    int r  = ch >> 2;              // 0..63
    int c8 = (ch & 3) << 3;        // 0,8,16,24
    aR[i] = r; aC8[i] = c8;
    int ml = m0 + r;
    long rowG = (long)rowOff + ((ml < Meff) ? ml : 0);   // clamp
    long ar = gidx ? (long)gidx[rowG] : rowG;
    aPtr[i] = A + aOff + ar * lda + c8;
  }
  const _Float16* bPtr[2]; int bR[2], bC8[2];
  #pragma unroll
  for (int i = 0; i < 2; ++i) {
    int ch = tid + i * 128;
    int n  = ch >> 2;
    int c8 = (ch & 3) << 3;
    bR[i] = n; bC8[i] = c8;
    int nl = n0 + n;
    int nc = (nl < N) ? nl : (N - 1);                    // clamp
    bPtr[i] = BT + bOff + (long)nc * ldb + c8;
  }

  auto loadT = [&](int k0, v8h* va, v8h* vb) {
    #pragma unroll
    for (int i = 0; i < 2; ++i) va[i] = *(const v8h*)(aPtr[i] + k0);
    #pragma unroll
    for (int i = 0; i < 2; ++i) vb[i] = *(const v8h*)(bPtr[i] + k0);
  };
  auto storeT = [&](int pb, v8h* va, v8h* vb) {
    #pragma unroll
    for (int i = 0; i < 2; ++i) *(v8h*)&As[pb][aR[i]][aC8[i]] = va[i];
    #pragma unroll
    for (int i = 0; i < 2; ++i) *(v8h*)&Bs[pb][bR[i]][bC8[i]] = vb[i];
  };

  v8f zero = {};
  v8f acc[2][2];
  acc[0][0] = zero; acc[0][1] = zero; acc[1][0] = zero; acc[1][1] = zero;

  int fr = lane & 15;
  int hi = lane >> 4;

  int nKT = K >> 5;                 // K % 32 == 0 guaranteed
  v8h va[2], vb[2];
  loadT(0, va, vb);
  for (int kt = 0; kt < nKT; ++kt) {
    int pb = kt & 1;
    storeT(pb, va, vb);
    __syncthreads();
    if (kt + 1 < nKT) loadT((kt + 1) << 5, va, vb);  // prefetch overlaps WMMA

    v16h afr[2], bfr[2];
    #pragma unroll
    for (int i = 0; i < 2; ++i) {
      int R = wm + i * 16 + fr;
      v8h a0 = *(const v8h*)&As[pb][R][hi * 8];
      v8h a1 = *(const v8h*)&As[pb][R][16 + hi * 8];
      afr[i] = __builtin_shufflevector(a0, a1, 0,1,2,3,4,5,6,7,8,9,10,11,12,13,14,15);
    }
    #pragma unroll
    for (int i = 0; i < 2; ++i) {
      int Cb = wn + i * 16 + fr;
      v8h b0 = *(const v8h*)&Bs[pb][Cb][hi * 8];
      v8h b1 = *(const v8h*)&Bs[pb][Cb][16 + hi * 8];
      bfr[i] = __builtin_shufflevector(b0, b1, 0,1,2,3,4,5,6,7,8,9,10,11,12,13,14,15);
    }
    #pragma unroll
    for (int i = 0; i < 2; ++i)
      #pragma unroll
      for (int j = 0; j < 2; ++j)
        acc[i][j] = __builtin_amdgcn_wmma_f32_16x16x32_f16(
            false, afr[i], false, bfr[j], (short)0, acc[i][j], false, false);
    // no trailing barrier: double-buffered LDS
  }

  // ---- store D: f32 16x16 C/D layout (vgpr v -> M=v / M=8+v) ----
  int sn = lane & 15;
  int mh = lane >> 4;
  #pragma unroll
  for (int j = 0; j < 2; ++j) {
    int nl = n0 + wn + j * 16 + sn;
    if (nl >= N) continue;
    float bv = bias ? bias[biOff + nl] : 0.f;
    #pragma unroll
    for (int i = 0; i < 2; ++i) {
      #pragma unroll
      for (int vr = 0; vr < 8; ++vr) {
        int ml = m0 + wm + i * 16 + mh * 8 + vr;
        if (ml < Meff) {
          float v = acc[i][j][vr] + bv;
          if (act == 1) v = (v > 0.f) ? v : 0.2f * v;
          long rowG = (long)rowOff + ml;
          if (C)   C[cOff + rowG * ldc + nl] = v;
          if (C16) C16[c16Off + rowG * ldc16 + nl] = (_Float16)v;
        }
      }
    }
  }
}

// ---------------------------------------------------------------------------
// Weight prep: tiled transpose + f32->f16 convert + zero-pad K to Kpad
// W: f32 [nb][d0][d1]  ->  T: f16 [nb][d1][Kpad],  T[z][n][k] = W[z][k][n]
// ---------------------------------------------------------------------------
__global__ void k_wt(const float* __restrict__ W, long wStride,
                     _Float16* __restrict__ T, long tStride,
                     int d0, int d1, int Kpad) {
  __shared__ float tile[32][33];
  const float* Wz = W + (long)blockIdx.z * wStride;
  _Float16*    Tz = T + (long)blockIdx.z * tStride;
  int kb = blockIdx.y * 32, nb = blockIdx.x * 32;
  for (int i = threadIdx.y; i < 32; i += 8) {
    int k = kb + i, n = nb + threadIdx.x;
    tile[i][threadIdx.x] = (k < d0 && n < d1) ? Wz[(long)k * d1 + n] : 0.f;
  }
  __syncthreads();
  for (int i = threadIdx.y; i < 32; i += 8) {
    int n = nb + i, k = kb + threadIdx.x;
    if (n < d1 && k < Kpad) Tz[(long)n * Kpad + k] = (_Float16)tile[threadIdx.x][i];
  }
}
// straight f32 -> f16 convert (for already n-major GRU weights)
__global__ void k_wcvt(const float* __restrict__ W, _Float16* __restrict__ T, long n) {
  long i = (long)blockIdx.x * blockDim.x + threadIdx.x;
  long s = (long)gridDim.x * blockDim.x;
  for (; i < n; i += s) T[i] = (_Float16)W[i];
}

// ---------------------------------------------------------------------------
// Elementwise / glue kernels
// ---------------------------------------------------------------------------
__global__ void k_zero_f(float* p, long n) {
  long i = (long)blockIdx.x * blockDim.x + threadIdx.x;
  long s = (long)gridDim.x * blockDim.x;
  for (; i < n; i += s) p[i] = 0.f;
}
__global__ void k_zero_i(int* p, int n) {
  int i = blockIdx.x * blockDim.x + threadIdx.x;
  if (i < n) p[i] = 0;
}
__global__ void k_cmd(const float* __restrict__ gt, int* __restrict__ cmds,
                      int* __restrict__ counts) {
  int n = blockIdx.x * blockDim.x + threadIdx.x;
  if (n >= NR) return;
  const float* r = gt + (long)n * TLc;
  float best = r[0]; int bi = 0;
  #pragma unroll
  for (int i = 1; i < Ff; ++i) { float v = r[i]; if (v > best) { best = v; bi = i; } }
  cmds[n] = bi;
  atomicAdd(&counts[bi], 1);
}
__global__ void k_base(const int* counts, int* baseb, int* cursor) {
  if (threadIdx.x == 0 && blockIdx.x == 0) {
    int s = 0;
    for (int f = 0; f < Ff; ++f) { baseb[f] = s; cursor[f] = s; s += counts[f]; }
  }
}
__global__ void k_scatter(const int* __restrict__ cmds, int* __restrict__ cursor,
                          int* __restrict__ pos, int* __restrict__ gidx) {
  int n = blockIdx.x * blockDim.x + threadIdx.x;
  if (n >= NR) return;
  int f = cmds[n];
  int p = atomicAdd(&cursor[f], 1);
  pos[n] = p;
  gidx[p] = n;
}
// f16 X0[n, 0:464]=inp_seq, [464:467]=bb, [467:471]=one_hot, [471:480]=0
__global__ void k_concat16(const float* __restrict__ inp, const float* __restrict__ bb,
                           const int* __restrict__ hier, _Float16* __restrict__ X0) {
  int n = blockIdx.x;
  int b = n / Tt;
  _Float16* o = X0 + (long)n * 544;
  for (int c = threadIdx.x; c < 480; c += blockDim.x) {
    float v;
    if (c < 464)      v = inp[(long)n * TLc + c];
    else if (c < 467) v = bb[b * 3 + (c - 464)];
    else if (c < 471) v = (hier[b] == (c - 467)) ? 1.f : 0.f;
    else              v = 0.f;
    o[c] = (_Float16)v;
  }
}
// h0 = code (f32 state + f16 GEMM operand)
__global__ void k_h0(const float* __restrict__ code, float* __restrict__ h,
                     _Float16* __restrict__ hh16) {
  int i = blockIdx.x * blockDim.x + threadIdx.x;
  if (i >= Bb * Hh) return;
  float v = code[i];
  h[i] = v;
  hh16[i] = (_Float16)v;
}
// one GRU step gate fusion; writes f32 state + f16 mirrors
__global__ void k_gru(const float* __restrict__ gi, const float* __restrict__ gh,
                      float* __restrict__ h, _Float16* __restrict__ hh16,
                      _Float16* __restrict__ g16, int t) {
  int i = blockIdx.x * blockDim.x + threadIdx.x;
  if (i >= Bb * Hh) return;
  int b = i >> 9, j = i & 511;
  long n = (long)b * Tt + t;
  float ir = gi[n * 1536 + j], iz = gi[n * 1536 + 512 + j], in = gi[n * 1536 + 1024 + j];
  float hr = gh[(long)b * 1536 + j], hz = gh[(long)b * 1536 + 512 + j], hn = gh[(long)b * 1536 + 1024 + j];
  float r  = 1.f / (1.f + expf(-(ir + hr)));
  float zg = 1.f / (1.f + expf(-(iz + hz)));
  float nn = tanhf(in + r * hn);
  float hv = (1.f - zg) * nn + zg * h[i];
  h[i] = hv;
  hh16[i] = (_Float16)hv;
  g16[n * 512 + j] = (_Float16)hv;
}
// f16 xf[pos[n], :] = [g16[n] | gt[n,16+f*28:+16] | bb[b] | 0-pad] (width 544)
__global__ void k_xf16(const _Float16* __restrict__ g16, const float* __restrict__ gt,
                       const float* __restrict__ bb, const int* __restrict__ cmds,
                       const int* __restrict__ pos, _Float16* __restrict__ xf) {
  int n = blockIdx.x;
  int f = cmds[n];
  long r = pos[n];
  int b = n / Tt;
  _Float16* o = xf + r * 544;
  for (int c = threadIdx.x; c < 544; c += blockDim.x) {
    _Float16 v;
    if (c < 512)      v = g16[(long)n * 512 + c];
    else if (c < 528) v = (_Float16)gt[(long)n * TLc + 16 + f * 28 + (c - 512)];
    else if (c < 531) v = (_Float16)bb[b * 3 + (c - 528)];
    else              v = (_Float16)0.f;
    o[c] = v;
  }
}
__global__ void k_denc16(const _Float16* __restrict__ g16, const float* __restrict__ code,
                         _Float16* __restrict__ denc) {
  int n = blockIdx.x;
  int b = n / Tt;
  _Float16* o = denc + (long)n * 1024;
  for (int c = threadIdx.x; c < 1024; c += blockDim.x)
    o[c] = (c < 512) ? g16[(long)n * 512 + c] : (_Float16)code[(long)b * 512 + (c - 512)];
}
// final assembly of `out`: func head cols 0..15 + selected-func 28-col block
__global__ void k_out(const float* __restrict__ funco, const float* __restrict__ d3b,
                      const float* __restrict__ f3b, const float* __restrict__ b3b,
                      const int* __restrict__ cmds, const int* __restrict__ pos,
                      float* __restrict__ out) {
  int n = blockIdx.x;
  int f = cmds[n];
  long r = pos[n];
  float* o = out + (long)n * TLc;
  int t = threadIdx.x;
  if (t < 16) o[t] = funco[(long)n * 16 + t];
  int base = 16 + f * 28;
  if (t < 8)        o[base + t] = d3b[r * 8 + t];
  else if (t < 16)  o[base + t] = d3b[(long)NR * 8 + r * 8 + (t - 8)];
  else if (t < 24)  o[base + t] = f3b[r * 8 + (t - 16)];
  else if (t < 28)  o[base + t] = b3b[r * 4 + (t - 24)];
}

// ---------------------------------------------------------------------------
// Host side
// ---------------------------------------------------------------------------
static inline void gemm16(hipStream_t s,
    const _Float16* A, long lda, long aF, long aP,
    const _Float16* BT, long ldb, long bF, long bP,
    const float* bias, long biF, long biP,
    float* C, long ldc, long cF, long cP,
    _Float16* C16, long ldc16, long c16F, long c16P,
    int M, int N, int K, int act, int nF, int nZ,
    const int* cnt, const int* base, const int* gidx)
{
  dim3 g((unsigned)((N + 63) / 64), (unsigned)((M + 63) / 64), (unsigned)nZ);
  k_gemm16<<<g, 128, 0, s>>>(A, lda, aF, aP, BT, ldb, bF, bP, bias, biF, biP,
                             C, ldc, cF, cP, C16, ldc16, c16F, c16P,
                             M, N, K, act, nF, cnt, base, gidx);
}
static inline void wt(hipStream_t s, const float* W, long wStride,
                      _Float16* T, long tStride, int d0, int d1, int Kpad, int nb) {
  dim3 g((unsigned)((d1 + 31) / 32), (unsigned)((Kpad + 31) / 32), (unsigned)nb);
  k_wt<<<g, dim3(32, 8), 0, s>>>(W, wStride, T, tStride, d0, d1, Kpad);
}

extern "C" void kernel_launch(void* const* d_in, const int* in_sizes, int n_in,
                              void* d_out, int out_size, void* d_ws, size_t ws_size,
                              hipStream_t stream) {
  (void)in_sizes; (void)n_in; (void)out_size; (void)ws_size;
  const float* inp_seq = (const float*)d_in[0];
  const float* code    = (const float*)d_in[1];
  const float* bbd     = (const float*)d_in[2];
  const int*   hier    = (const int*)d_in[3];
  const float* gt      = (const float*)d_in[4];
  const float* iW1 = (const float*)d_in[5],  *ib1 = (const float*)d_in[6];
  const float* iW2 = (const float*)d_in[7],  *ib2 = (const float*)d_in[8];
  const float* iW3 = (const float*)d_in[9],  *ib3 = (const float*)d_in[10];
  const float* Wih = (const float*)d_in[11], *Whh = (const float*)d_in[12];
  const float* bih = (const float*)d_in[13], *bhh = (const float*)d_in[14];
  const float* cW1 = (const float*)d_in[15], *cb1 = (const float*)d_in[16];
  const float* cW2 = (const float*)d_in[17], *cb2 = (const float*)d_in[18];
  const float* cW3 = (const float*)d_in[19], *cb3 = (const float*)d_in[20];
  const float* nW1 = (const float*)d_in[21], *nb1 = (const float*)d_in[22];
  const float* nW2 = (const float*)d_in[23], *nb2 = (const float*)d_in[24];
  const float* nW3 = (const float*)d_in[25], *nb3 = (const float*)d_in[26];
  const float* uW1 = (const float*)d_in[27], *ub1 = (const float*)d_in[28];
  const float* uW2 = (const float*)d_in[29], *ub2 = (const float*)d_in[30];
  const float* uW3 = (const float*)d_in[31], *ub3 = (const float*)d_in[32];
  const float* dW1 = (const float*)d_in[33], *db1 = (const float*)d_in[34];
  const float* dW2 = (const float*)d_in[35], *db2 = (const float*)d_in[36];
  const float* dW3 = (const float*)d_in[37], *db3 = (const float*)d_in[38];
  const float* fW1 = (const float*)d_in[39], *fb1 = (const float*)d_in[40];
  const float* fW2 = (const float*)d_in[41], *fb2 = (const float*)d_in[42];
  const float* fW3 = (const float*)d_in[43], *fb3 = (const float*)d_in[44];
  const float* bW1 = (const float*)d_in[45], *bb1 = (const float*)d_in[46];
  const float* bW2 = (const float*)d_in[47], *bb2 = (const float*)d_in[48];
  const float* bW3 = (const float*)d_in[49], *bb3 = (const float*)d_in[50];

  char* arena = (char*)d_ws;
  size_t off = 0;
  auto aF32 = [&](size_t n) { float* p = (float*)(arena + off);
    off = (off + n * 4 + 255) & ~(size_t)255; return p; };
  auto aH16 = [&](size_t n) { _Float16* p = (_Float16*)(arena + off);
    off = (off + n * 2 + 255) & ~(size_t)255; return p; };
  auto aI32 = [&](size_t n) { int* p = (int*)(arena + off);
    off = (off + n * 4 + 255) & ~(size_t)255; return p; };

  // f16 weight mirrors (n-major [N][Kpad])
  _Float16* TiW1 = aH16((size_t)512 * 480);
  _Float16* TiW2 = aH16((size_t)512 * 512);
  _Float16* TiW3 = aH16((size_t)512 * 512);
  _Float16* TWih = aH16((size_t)1536 * 512);
  _Float16* TWhh = aH16((size_t)1536 * 512);
  _Float16* TuW1 = aH16((size_t)256 * 512);
  _Float16* TuW2 = aH16((size_t)128 * 256);
  _Float16* TuW3 = aH16((size_t)16 * 128);
  _Float16* TdW1 = aH16((size_t)32 * 256 * 512);
  _Float16* TdW2 = aH16((size_t)32 * 128 * 256);
  _Float16* TdW3 = aH16((size_t)32 * 8 * 128);
  _Float16* TfW1 = aH16((size_t)16 * 256 * 544);
  _Float16* TfW2 = aH16((size_t)16 * 128 * 256);
  _Float16* TfW3 = aH16((size_t)16 * 8 * 128);
  _Float16* TbW1 = aH16((size_t)16 * 256 * 512);
  _Float16* TbW2 = aH16((size_t)16 * 128 * 256);
  _Float16* TbW3 = aH16((size_t)16 * 4 * 128);
  _Float16* TcW1 = aH16((size_t)256 * 1024);
  _Float16* TcW2 = aH16((size_t)64 * 256);
  _Float16* TcW3 = aH16((size_t)4 * 64);
  _Float16* TnW1 = aH16((size_t)512 * 1024);
  _Float16* TnW2 = aH16((size_t)512 * 512);
  _Float16* TnW3 = aH16((size_t)2048 * 512);
  // f16 activations
  _Float16* X016 = aH16((size_t)NR * 544);     // concat input; reused as xf
  _Float16* a1   = aH16((size_t)NR * 512);     // mlp tmp; reused by nc
  _Float16* a2   = aH16((size_t)NR * 512);
  _Float16* xb16 = aH16((size_t)NR * 512);
  _Float16* g16  = aH16((size_t)NR * 512);
  _Float16* hh16 = aH16((size_t)Bb * 512);
  _Float16* den6 = aH16((size_t)NR * 1024);
  _Float16* u116 = aH16((size_t)NR * 256);
  _Float16* u216 = aH16((size_t)NR * 128);
  _Float16* d1b6 = aH16((size_t)2 * NR * 256);
  _Float16* d2b6 = aH16((size_t)2 * NR * 128);
  _Float16* f1b6 = aH16((size_t)NR * 256);
  _Float16* f2b6 = aH16((size_t)NR * 128);
  _Float16* b1b6 = aH16((size_t)NR * 256);
  _Float16* b2b6 = aH16((size_t)NR * 128);
  _Float16* c116 = aH16((size_t)NR * 256);
  _Float16* c216 = aH16((size_t)NR * 64);
  // f32 buffers
  float* gi  = aF32((size_t)NR * 1536);
  float* gh  = aF32((size_t)Bb * 1536);
  float* hf  = aF32((size_t)Bb * 512);
  float* fo  = aF32((size_t)NR * 16);
  float* d3b = aF32((size_t)2 * NR * 8);
  float* f3b = aF32((size_t)NR * 8);
  float* b3b = aF32((size_t)NR * 4);
  int* cmds   = aI32(NR);
  int* pos    = aI32(NR);
  int* gidx   = aI32(NR);
  int* counts = aI32(16);
  int* baseb  = aI32(16);
  int* cursor = aI32(16);

  float* outF = (float*)d_out;
  long   oNC  = (long)NR * TLc;
  long   oCH  = oNC + (long)NR * 2048;

  // ---- bucketing + output zero ----
  k_zero_f<<<512, 256, 0, stream>>>(outF, (long)NR * TLc);
  k_zero_i<<<1, 64, 0, stream>>>(counts, 16);
  k_cmd<<<NR / 256, 256, 0, stream>>>(gt, cmds, counts);
  k_base<<<1, 32, 0, stream>>>(counts, baseb, cursor);
  k_scatter<<<NR / 256, 256, 0, stream>>>(cmds, cursor, pos, gidx);

  // ---- weight prep: transpose/convert/pad to f16 ----
  wt(stream, iW1, 0, TiW1, 0, 471, 512, 480, 1);
  wt(stream, iW2, 0, TiW2, 0, 512, 512, 512, 1);
  wt(stream, iW3, 0, TiW3, 0, 512, 512, 512, 1);
  k_wcvt<<<512, 256, 0, stream>>>(Wih, TWih, (long)1536 * 512);
  k_wcvt<<<512, 256, 0, stream>>>(Whh, TWhh, (long)1536 * 512);
  wt(stream, uW1, 0, TuW1, 0, 512, 256, 512, 1);
  wt(stream, uW2, 0, TuW2, 0, 256, 128, 256, 1);
  wt(stream, uW3, 0, TuW3, 0, 128, 16, 128, 1);
  wt(stream, dW1, (long)512 * 256, TdW1, (long)256 * 512, 512, 256, 512, 32);
  wt(stream, dW2, (long)256 * 128, TdW2, (long)128 * 256, 256, 128, 256, 32);
  wt(stream, dW3, (long)128 * 8,   TdW3, (long)8 * 128,   128, 8, 128, 32);
  wt(stream, fW1, (long)531 * 256, TfW1, (long)256 * 544, 531, 256, 544, 16);
  wt(stream, fW2, (long)256 * 128, TfW2, (long)128 * 256, 256, 128, 256, 16);
  wt(stream, fW3, (long)128 * 8,   TfW3, (long)8 * 128,   128, 8, 128, 16);
  wt(stream, bW1, (long)512 * 256, TbW1, (long)256 * 512, 512, 256, 512, 16);
  wt(stream, bW2, (long)256 * 128, TbW2, (long)128 * 256, 256, 128, 256, 16);
  wt(stream, bW3, (long)128 * 4,   TbW3, (long)4 * 128,   128, 4, 128, 16);
  wt(stream, cW1, 0, TcW1, 0, 1024, 256, 1024, 1);
  wt(stream, cW2, 0, TcW2, 0, 256, 64, 256, 1);
  wt(stream, cW3, 0, TcW3, 0, 64, 4, 64, 1);
  wt(stream, nW1, 0, TnW1, 0, 1024, 512, 1024, 1);
  wt(stream, nW2, 0, TnW2, 0, 512, 512, 512, 1);
  wt(stream, nW3, 0, TnW3, 0, 512, 2048, 512, 1);

  // ---- input MLP: 480(pad) -> 512 -> 512 -> 512 ----
  k_concat16<<<NR, 128, 0, stream>>>(inp_seq, bbd, hier, X016);
  gemm16(stream, X016, 544, 0, 0, TiW1, 480, 0, 0, ib1, 0, 0,
         0, 0, 0, 0, a1, 512, 0, 0, NR, 512, 480, 1, 1, 1, 0, 0, 0);
  gemm16(stream, a1, 512, 0, 0, TiW2, 512, 0, 0, ib2, 0, 0,
         0, 0, 0, 0, a2, 512, 0, 0, NR, 512, 512, 1, 1, 1, 0, 0, 0);
  gemm16(stream, a2, 512, 0, 0, TiW3, 512, 0, 0, ib3, 0, 0,
         0, 0, 0, 0, xb16, 512, 0, 0, NR, 512, 512, 0, 1, 1, 0, 0, 0);

  // ---- GRU ----
  gemm16(stream, xb16, 512, 0, 0, TWih, 512, 0, 0, bih, 0, 0,
         gi, 1536, 0, 0, 0, 0, 0, 0, NR, 1536, 512, 0, 1, 1, 0, 0, 0);
  k_h0<<<(Bb * Hh) / 256, 256, 0, stream>>>(code, hf, hh16);
  for (int t = 0; t < Tt; ++t) {
    gemm16(stream, hh16, 512, 0, 0, TWhh, 512, 0, 0, bhh, 0, 0,
           gh, 1536, 0, 0, 0, 0, 0, 0, Bb, 1536, 512, 0, 1, 1, 0, 0, 0);
    k_gru<<<(Bb * Hh) / 256, 256, 0, stream>>>(gi, gh, hf, hh16, g16, t);
  }

  // ---- shared func head: 512 -> 256 -> 128 -> 16 ----
  gemm16(stream, g16, 512, 0, 0, TuW1, 512, 0, 0, ub1, 0, 0,
         0, 0, 0, 0, u116, 256, 0, 0, NR, 256, 512, 1, 1, 1, 0, 0, 0);
  gemm16(stream, u116, 256, 0, 0, TuW2, 256, 0, 0, ub2, 0, 0,
         0, 0, 0, 0, u216, 128, 0, 0, NR, 128, 256, 1, 1, 1, 0, 0, 0);
  gemm16(stream, u216, 128, 0, 0, TuW3, 128, 0, 0, ub3, 0, 0,
         fo, 16, 0, 0, 0, 0, 0, 0, NR, 16, 128, 0, 1, 1, 0, 0, 0);

  // ---- disc heads, bucketed batched over (f,p): z = p*16 + f ----
  gemm16(stream, g16, 512, 0, 0, TdW1, 512, (long)2 * 256 * 512, (long)256 * 512,
         db1, 512, 256, 0, 0, 0, 0, d1b6, 256, 0, (long)NR * 256,
         NR, 256, 512, 1, 16, 32, counts, baseb, gidx);
  gemm16(stream, d1b6, 256, 0, (long)NR * 256, TdW2, 256, (long)2 * 128 * 256, (long)128 * 256,
         db2, 256, 128, 0, 0, 0, 0, d2b6, 128, 0, (long)NR * 128,
         NR, 128, 256, 1, 16, 32, counts, baseb, 0);
  gemm16(stream, d2b6, 128, 0, (long)NR * 128, TdW3, 128, (long)2 * 8 * 128, (long)8 * 128,
         db3, 16, 8, d3b, 8, 0, (long)NR * 8, 0, 0, 0, 0,
         NR, 8, 128, 0, 16, 32, counts, baseb, 0);

  // ---- float heads: compact xf (544 pad) -> 256 -> 128 -> 8 ----
  k_xf16<<<NR, 128, 0, stream>>>(g16, gt, bbd, cmds, pos, X016);
  gemm16(stream, X016, 544, 0, 0, TfW1, 544, (long)256 * 544, 0, fb1, 256, 0,
         0, 0, 0, 0, f1b6, 256, 0, 0, NR, 256, 544, 1, 16, 16, counts, baseb, 0);
  gemm16(stream, f1b6, 256, 0, 0, TfW2, 256, (long)128 * 256, 0, fb2, 128, 0,
         0, 0, 0, 0, f2b6, 128, 0, 0, NR, 128, 256, 1, 16, 16, counts, baseb, 0);
  gemm16(stream, f2b6, 128, 0, 0, TfW3, 128, (long)8 * 128, 0, fb3, 8, 0,
         f3b, 8, 0, 0, 0, 0, 0, 0, NR, 8, 128, 0, 16, 16, counts, baseb, 0);

  // ---- bool heads: 512 -> 256 -> 128 -> 4 (gathered) ----
  gemm16(stream, g16, 512, 0, 0, TbW1, 512, (long)256 * 512, 0, bb1, 256, 0,
         0, 0, 0, 0, b1b6, 256, 0, 0, NR, 256, 512, 1, 16, 16, counts, baseb, gidx);
  gemm16(stream, b1b6, 256, 0, 0, TbW2, 256, (long)128 * 256, 0, bb2, 128, 0,
         0, 0, 0, 0, b2b6, 128, 0, 0, NR, 128, 256, 1, 16, 16, counts, baseb, 0);
  gemm16(stream, b2b6, 128, 0, 0, TbW3, 128, (long)4 * 128, 0, bb3, 4, 0,
         b3b, 4, 0, 0, 0, 0, 0, 0, NR, 4, 128, 0, 16, 16, counts, baseb, 0);

  // ---- child + next_codes from double_enc = [gru_out | code] ----
  k_denc16<<<NR, 128, 0, stream>>>(g16, code, den6);
  gemm16(stream, den6, 1024, 0, 0, TcW1, 1024, 0, 0, cb1, 0, 0,
         0, 0, 0, 0, c116, 256, 0, 0, NR, 256, 1024, 1, 1, 1, 0, 0, 0);
  gemm16(stream, c116, 256, 0, 0, TcW2, 256, 0, 0, cb2, 0, 0,
         0, 0, 0, 0, c216, 64, 0, 0, NR, 64, 256, 1, 1, 1, 0, 0, 0);
  gemm16(stream, c216, 64, 0, 0, TcW3, 64, 0, 0, cb3, 0, 0,
         outF + oCH, 4, 0, 0, 0, 0, 0, 0, NR, 4, 64, 0, 1, 1, 0, 0, 0);

  gemm16(stream, den6, 1024, 0, 0, TnW1, 1024, 0, 0, nb1, 0, 0,
         0, 0, 0, 0, a1, 512, 0, 0, NR, 512, 1024, 1, 1, 1, 0, 0, 0);
  gemm16(stream, a1, 512, 0, 0, TnW2, 512, 0, 0, nb2, 0, 0,
         0, 0, 0, 0, a2, 512, 0, 0, NR, 512, 512, 1, 1, 1, 0, 0, 0);
  gemm16(stream, a2, 512, 0, 0, TnW3, 512, 0, 0, nb3, 0, 0,
         outF + oNC, 2048, 0, 0, 0, 0, 0, 0, NR, 2048, 512, 0, 1, 1, 0, 0, 0);

  // ---- assemble `out` ----
  k_out<<<NR, 32, 0, stream>>>(fo, d3b, f3b, b3b, cmds, pos, outF);
}
